// YoloLoss_6622839571268
// MI455X (gfx1250) — compile-verified, matrix operand order checked by reading, not run on hardware
//
#include <hip/hip_runtime.h>
#include <hip/hip_bf16.h>

typedef __attribute__((ext_vector_type(16))) _Float16 v16h;
typedef __attribute__((ext_vector_type(8)))  float    v8f;

#define NAV 3
#define NCV 80
#define NMATCH 1024
#define BSZ 16

// workspace layout (float offsets)
#define TOBJ0_OFF 0
#define TOBJ1_OFF 19200              // 16*3*20*20
#define TOBJ2_OFF (19200 + 76800)    // + 16*3*40*40
#define BOX_OFF   (19200 + 76800 + 307200)  // 403200
#define CLS_OFF   (BOX_OFF + 256)
#define OBJ0_OFF  (CLS_OFF + 256)
#define OBJ1_OFF  (OBJ0_OFF + 256)
#define OBJ2_OFF  (OBJ1_OFF + 256)
#define WS_FLOATS (OBJ2_OFF + 256)   // 404480 floats ~ 1.62 MB

__device__ __forceinline__ float softplus_stable(float x) {
    // max(x,0) + log1p(exp(-|x|))  == BCEWithLogits(x, t=0)
    return fmaxf(x, 0.0f) + log1pf(expf(-fabsf(x)));
}

__global__ void zero_kernel(float* __restrict__ ws, int n) {
    int i = blockIdx.x * blockDim.x + threadIdx.x;
    int stride = gridDim.x * blockDim.x;
    for (; i < n; i += stride) ws[i] = 0.0f;
}

// One thread per match. 4 blocks * 256 threads = 1024 matches.
__global__ void match_kernel(const float* __restrict__ p,
                             const float* __restrict__ tbox,
                             const float* __restrict__ anch,
                             const int*   __restrict__ bi,
                             const int*   __restrict__ ai,
                             const int*   __restrict__ gj,
                             const int*   __restrict__ gi,
                             const int*   __restrict__ tcls,
                             float* __restrict__ tobj,      // this scale's tobj buffer
                             float* __restrict__ box_part,  // 256-slot region
                             float* __restrict__ cls_part,  // 256-slot region
                             int h, int part_base) {
    const int t   = threadIdx.x;
    const int tid = blockIdx.x * blockDim.x + t;   // < 1024
    const int hw  = h * h;

    const int bb = bi[tid];
    const int aa = ai[tid];
    const int gjv = gj[tid];
    const int giv = gi[tid];
    const int tc  = tcls[tid];

    // pred[b, a, gj, gi, c] = p[((b*255 + a*85 + c)*h + gj)*h + gi]
    const int base = ((bb * (NAV * 85) + aa * 85) * h + gjv) * h + giv;

    const float px = p[base];
    const float py = p[base + hw];
    const float pw = p[base + 2 * hw];
    const float ph = p[base + 3 * hw];

    const float sx = 1.0f / (1.0f + expf(-px));
    const float sy = 1.0f / (1.0f + expf(-py));
    const float sw = 1.0f / (1.0f + expf(-pw));
    const float sh = 1.0f / (1.0f + expf(-ph));

    const float bx = sx * 2.0f - 0.5f;
    const float by = sy * 2.0f - 0.5f;
    const float bw = (sw * 2.0f) * (sw * 2.0f) * anch[2 * tid];
    const float bh = (sh * 2.0f) * (sh * 2.0f) * anch[2 * tid + 1];

    const float fw = (float)h;
    const float tx = tbox[4 * tid]     * fw - (float)giv;
    const float ty = tbox[4 * tid + 1] * fw - (float)gjv;
    const float tw = tbox[4 * tid + 2] * fw;
    const float th = tbox[4 * tid + 3] * fw;

    // ---- CIoU (matches reference exactly, eps=1e-7 on y-extents/union/c2) ----
    const float eps = 1e-7f;
    const float b1x1 = bx - bw * 0.5f, b1x2 = bx + bw * 0.5f;
    const float b1y1 = by - bh * 0.5f, b1y2 = by + bh * 0.5f;
    const float b2x1 = tx - tw * 0.5f, b2x2 = tx + tw * 0.5f;
    const float b2y1 = ty - th * 0.5f, b2y2 = ty + th * 0.5f;

    const float inter = fmaxf(fminf(b1x2, b2x2) - fmaxf(b1x1, b2x1), 0.0f) *
                        fmaxf(fminf(b1y2, b2y2) - fmaxf(b1y1, b2y1), 0.0f);
    const float w1 = b1x2 - b1x1, h1 = b1y2 - b1y1 + eps;
    const float w2 = b2x2 - b2x1, h2 = b2y2 - b2y1 + eps;
    const float uni = w1 * h1 + w2 * h2 - inter + eps;
    const float iou = inter / uni;

    const float cw = fmaxf(b1x2, b2x2) - fminf(b1x1, b2x1);
    const float ch = fmaxf(b1y2, b2y2) - fminf(b1y1, b2y1);
    const float c2 = cw * cw + ch * ch + eps;
    const float dx = b2x1 + b2x2 - b1x1 - b1x2;
    const float dy = b2y1 + b2y2 - b1y1 - b1y2;
    const float rho2 = (dx * dx + dy * dy) * 0.25f;

    const float inv_pi2_4 = 0.405284734569351f;  // 4 / pi^2
    const float da = atanf(w2 / h2) - atanf(w1 / h1);
    const float v  = inv_pi2_4 * da * da;
    const float alpha = v / (v - iou + (1.0f + eps));
    const float ciou  = iou - (rho2 / c2 + v * alpha);

    // scatter tobj = clip(iou, 0): atomicMax on uint bits (order-preserving for
    // non-negative floats) -> deterministic resolution of duplicate indices.
    {
        const float tval = fmaxf(ciou, 0.0f);
        unsigned int* ta = (unsigned int*)&tobj[((bb * NAV + aa) * h + gjv) * h + giv];
        atomicMax(ta, __float_as_uint(tval));
    }

    // class BCE sum: sum softplus over 80 classes, minus logit at target class
    const int cbase = base + 5 * hw;
    float csum = 0.0f;
    #pragma unroll 4
    for (int c = 0; c < NCV; ++c) {
        csum += softplus_stable(p[cbase + c * hw]);
    }
    csum -= p[cbase + tc * hw];

    // block reduction of (1 - ciou) and csum
    __shared__ float sb[256];
    __shared__ float sc[256];
    sb[t] = 1.0f - ciou;
    sc[t] = csum;
    __syncthreads();
    for (int s = 128; s > 0; s >>= 1) {
        if (t < s) { sb[t] += sb[t + s]; sc[t] += sc[t + s]; }
        __syncthreads();
    }
    if (t == 0) {
        box_part[part_base + blockIdx.x] = sb[0];
        cls_part[part_base + blockIdx.x] = sc[0];
    }
}

// Coalesced pass over the objectness channel + tobj; 256 blocks x 256 threads.
__global__ void obj_kernel(const float* __restrict__ p,
                           const float* __restrict__ tobj,
                           float* __restrict__ part, int h) {
    const int t = threadIdx.x;
    const int n = BSZ * NAV * h * h;
    const int hw = h * h;
    float sum = 0.0f;
    for (int idx = blockIdx.x * blockDim.x + t; idx < n; idx += gridDim.x * blockDim.x) {
        const int giv = idx % h;
        int r = idx / h;
        const int gjv = r % h;  r /= h;
        const int aa  = r % NAV;
        const int bb  = r / NAV;
        const float x = __builtin_nontemporal_load(
            &p[((bb * (NAV * 85) + aa * 85 + 4) * h + gjv) * h + giv]);
        const float tv = tobj[idx];
        sum += fmaxf(x, 0.0f) - x * tv + log1pf(expf(-fabsf(x)));
    }
    __shared__ float sb[256];
    sb[t] = sum;
    __syncthreads();
    for (int s = 128; s > 0; s >>= 1) {
        if (t < s) sb[t] += sb[t + s];
        __syncthreads();
    }
    if (t == 0) part[blockIdx.x] = sb[0];
}

// Single wave32. Five weighted 256-element partial regions reduced on the
// matrix pipe: values are split into exact f16 (hi,lo) pairs and multiplied by
// an all-ones B via v_wmma_f32_16x16x32_f16 (f32 accumulate) -> row sums.
__global__ void final_reduce(const float* __restrict__ ws, float* __restrict__ out) {
    const int lane = threadIdx.x;   // 0..31
    const int m = lane & 15;

    v16h ones;
    #pragma unroll
    for (int i = 0; i < 16; ++i) ones[i] = (_Float16)1.0f;

    const int   offs[5] = {BOX_OFF, CLS_OFF, OBJ0_OFF, OBJ1_OFF, OBJ2_OFF};
    const float wgts[5] = {
        0.05f / (float)NMATCH,                    // BOX_RATIO / n_match (same all scales)
        0.5f  / ((float)NMATCH * (float)NCV),     // CLS_RATIO / (n_match*NC)
        0.4f  / 19200.0f,                         // balance0 / (16*3*20*20)
        1.0f  / 76800.0f,                         // balance1 / (16*3*40*40)
        4.0f  / 307200.0f                         // balance2 / (16*3*80*80)
    };

    v8f acc = {};
    #pragma unroll
    for (int r = 0; r < 5; ++r) {
        const float* src = ws + offs[r];
        const float wgt = wgts[r];
        v16h av;
        #pragma unroll
        for (int q = 0; q < 8; ++q) {
            // A-matrix 16-bit 16x32 layout: lanes 0-15 hold K=0..7,16..23 for M=lane;
            // lanes 16-31 hold K=8..15,24..31 for M=lane-16. Each VGPR = one (hi,lo) pair.
            const int pidx = (lane < 16) ? ((q < 4) ? q : q + 4)
                                         : ((q < 4) ? q + 4 : q + 8);
            const float val = src[pidx * 16 + m] * wgt;
            const _Float16 hi = (_Float16)val;
            const _Float16 lo = (_Float16)(val - (float)hi);
            av[2 * q]     = hi;
            av[2 * q + 1] = lo;
        }
        acc = __builtin_amdgcn_wmma_f32_16x16x32_f16(
            /*neg_a=*/false, av, /*neg_b=*/false, ones,
            /*c_mod=*/(short)0, acc, /*reuse_a=*/false, /*reuse_b=*/false);
    }

    // D[m][n] = rowsum_m (same for every column n). Lane n<16 holds M=0..7 in
    // the 8 acc VGPRs, lane n+16 holds M=8..15.
    float s = acc[0] + acc[1] + acc[2] + acc[3] + acc[4] + acc[5] + acc[6] + acc[7];
    float total = __shfl(s, 0, 32) + __shfl(s, 16, 32);
    if (lane == 0) out[0] = total;
}

extern "C" void kernel_launch(void* const* d_in, const int* in_sizes, int n_in,
                              void* d_out, int out_size, void* d_ws, size_t ws_size,
                              hipStream_t stream) {
    (void)in_sizes; (void)n_in; (void)out_size; (void)ws_size;
    float* ws = (float*)d_ws;

    const int hs[3] = {20, 40, 80};
    const int tobj_off[3] = {TOBJ0_OFF, TOBJ1_OFF, TOBJ2_OFF};
    const int obj_off[3]  = {OBJ0_OFF, OBJ1_OFF, OBJ2_OFF};

    zero_kernel<<<512, 256, 0, stream>>>(ws, WS_FLOATS);

    for (int i = 0; i < 3; ++i) {
        const float* p    = (const float*)d_in[8 * i + 0];
        const float* tbox = (const float*)d_in[8 * i + 1];
        const float* anch = (const float*)d_in[8 * i + 2];
        const int*   bi   = (const int*)  d_in[8 * i + 3];
        const int*   ai   = (const int*)  d_in[8 * i + 4];
        const int*   gj   = (const int*)  d_in[8 * i + 5];
        const int*   gi   = (const int*)  d_in[8 * i + 6];
        const int*   tc   = (const int*)  d_in[8 * i + 7];
        match_kernel<<<NMATCH / 256, 256, 0, stream>>>(
            p, tbox, anch, bi, ai, gj, gi, tc,
            ws + tobj_off[i], ws + BOX_OFF, ws + CLS_OFF, hs[i], i * (NMATCH / 256));
    }

    for (int i = 0; i < 3; ++i) {
        const float* p = (const float*)d_in[8 * i + 0];
        obj_kernel<<<256, 256, 0, stream>>>(p, ws + tobj_off[i], ws + obj_off[i], hs[i]);
    }

    final_reduce<<<1, 32, 0, stream>>>(ws, (float*)d_out);
}
